// MultiHeadAttention_60971355734022
// MI455X (gfx1250) — compile-verified, matrix operand order checked by reading, not run on hardware
//
#include <hip/hip_runtime.h>
#include <hip/hip_bf16.h>
#include <math.h>

// ---------------------------------------------------------------------------
// CDNA5 (gfx1250) WMMA multi-head attention.
//   D = softmax(causal( (xWq)(xWk)^T / sqrt(2048) )) (xWv) Wf + biases
// v_wmma_f32_16x16x32_f16 for all matrix math; async global->LDS staging
// (ASYNCcnt, instruction-offset addressing) with double-buffered LDS tiles.
// ---------------------------------------------------------------------------

typedef __attribute__((ext_vector_type(16))) _Float16 v16h;
typedef __attribute__((ext_vector_type(8)))  _Float16 v8h;
typedef __attribute__((ext_vector_type(8)))  float    v8f;
typedef __attribute__((ext_vector_type(4)))  float    v4f;

#define DMODEL 1024
#define NHEAD  16
#define HD     64
#define SLEN   2048
#define BATCH  4
#define MROWS  (BATCH * SLEN)   // 8192

union F16x16 { v16h v; v8h h[2]; };

__device__ __forceinline__ v8f vzero8() {
  v8f z;
#pragma unroll
  for (int i = 0; i < 8; ++i) z[i] = 0.0f;
  return z;
}

// WMMA operand fragment from a row-major f16 tile (A-layout; also B when the
// tile is stored [N][K]).  ISA 7.12.2: lane = hi*16 + r holds row r; elems
// 0..7 = K hi*8.., elems 8..15 = K 16+hi*8..  -> two 128-bit loads.
__device__ __forceinline__ v16h ld_frag(const _Float16* base, int stride_h) {
  const int lane = threadIdx.x & 31;
  const int r = lane & 15, hi = lane >> 4;
  const _Float16* p = base + r * stride_h + hi * 8;
  F16x16 f;
  f.h[0] = *(const v8h*)(p);
  f.h[1] = *(const v8h*)(p + 16);
  return f.v;
}

#define WMMA_F32_F16(A, B, C) \
  __builtin_amdgcn_wmma_f32_16x16x32_f16(false, (A), false, (B), (short)0, (C), false, false)

// ---- gfx1250 async global->LDS copy (ASYNCcnt), 16B per lane --------------
// ISA 8.2: INST_OFFSET is added to BOTH the LDS dest and the global source,
// so one (lds, global) base pair serves a whole contiguous multi-chunk copy.
template <int BYTE_OFF>
__device__ __forceinline__ void async_b128(const _Float16* g, _Float16* l) {
  asm volatile("global_load_async_to_lds_b128 %0, %1, off offset:%2"
               :: "v"((unsigned int)(uintptr_t)l),
                  "v"((unsigned long long)(uintptr_t)g),
                  "i"(BYTE_OFF)
               : "memory");
}
__device__ __forceinline__ void wait_async0() {
  asm volatile("s_wait_asynccnt 0" ::: "memory");
}

// ---------------------------------------------------------------------------
// Elementwise f32 -> f16 convert (q/k/v prep), 8 elems/thread.
// ---------------------------------------------------------------------------
__global__ __launch_bounds__(256)
void cvt_f32_f16(const float* __restrict__ src, _Float16* __restrict__ dst,
                 int n8) {
  const int i = blockIdx.x * 256 + threadIdx.x;
  if (i < n8) {
    const v4f* s = (const v4f*)(src + (size_t)i * 8);
    const v4f x0 = s[0], x1 = s[1];
    v8h o;
#pragma unroll
    for (int j = 0; j < 4; ++j) {
      o[j]     = (_Float16)x0[j];
      o[j + 4] = (_Float16)x1[j];
    }
    *(v8h*)(dst + (size_t)i * 8) = o;
  }
}

// ---------------------------------------------------------------------------
// Weight prep: Wt[n][k] = (f16) W[k][n]   (32x32 LDS tile transpose)
// ---------------------------------------------------------------------------
__global__ __launch_bounds__(256)
void transpose_f32_f16(const float* __restrict__ W, _Float16* __restrict__ Wt) {
  __shared__ float tile[32][33];
  const int n0 = blockIdx.x * 32, k0 = blockIdx.y * 32;
  const int tx = threadIdx.x & 31, ty = threadIdx.x >> 5;   // ty: 0..7
#pragma unroll
  for (int i = 0; i < 4; ++i)
    tile[ty + i * 8][tx] = W[(size_t)(k0 + ty + i * 8) * DMODEL + n0 + tx];
  __syncthreads();
#pragma unroll
  for (int i = 0; i < 4; ++i)
    Wt[(size_t)(n0 + ty + i * 8) * DMODEL + k0 + tx] =
        (_Float16)tile[tx][ty + i * 8];
}

// ---------------------------------------------------------------------------
// Tiled GEMM: C[M x 1024] = A[M x 1024] * W[1024 x 1024] + bias
// A and Wt both f16 (Wt pre-transposed [N][K]).  Block: 256 threads (8 waves),
// tile 128(M) x 128(N), K-step 32, double-buffered LDS filled by
// global_load_async_to_lds_b128 overlapping the WMMAs of the current tile.
// OUT_MODE 0: f16 head-major [B][H][S][64]; OUT_MODE 1: f32 row-major.
// ---------------------------------------------------------------------------
template <int OUT_MODE>
__global__ __launch_bounds__(256)
void gemm1024_wmma(const _Float16* __restrict__ A,
                   const _Float16* __restrict__ Wt,
                   const float* __restrict__ bias, void* __restrict__ outp) {
  __shared__ alignas(16) _Float16 As[2][128][40];  // [m][k]
  __shared__ alignas(16) _Float16 Bs[2][128][40];  // [n][k]

  const int t    = threadIdx.x;
  const int wave = t >> 5;
  const int lane = t & 31;
  const int r    = lane & 15, hi = lane >> 4;
  const int n0   = blockIdx.x * 128;
  const int m0   = blockIdx.y * 128;

  // Staging assignment: thread t -> tile row t>>1, 16-half chunk (t&1)*16.
  const int srow = t >> 1, scol = (t & 1) * 16;
  const _Float16* aptr = A  + (size_t)(m0 + srow) * DMODEL + scol;
  const _Float16* bptr = Wt + (size_t)(n0 + srow) * DMODEL + scol;

  v8f acc[8];
#pragma unroll
  for (int j = 0; j < 8; ++j) acc[j] = vzero8();

  // Prologue: async-stage K-tile 0 into buffer 0 (offset applies to both
  // the LDS dest and the global source).
  async_b128<0>(aptr, &As[0][srow][scol]);
  async_b128<16>(aptr, &As[0][srow][scol]);
  async_b128<0>(bptr, &Bs[0][srow][scol]);
  async_b128<16>(bptr, &Bs[0][srow][scol]);
  wait_async0();
  __syncthreads();

#pragma unroll 2
  for (int k0 = 0; k0 < DMODEL; k0 += 32) {
    const int buf = (k0 >> 5) & 1;
    const bool has_next = (k0 + 32 < DMODEL);
    if (has_next) {                    // kick off tile i+1 before the math
      const int nb = buf ^ 1;
      async_b128<0>(aptr + k0 + 32, &As[nb][srow][scol]);
      async_b128<16>(aptr + k0 + 32, &As[nb][srow][scol]);
      async_b128<0>(bptr + k0 + 32, &Bs[nb][srow][scol]);
      async_b128<16>(bptr + k0 + 32, &Bs[nb][srow][scol]);
      if (k0 + 64 < DMODEL) {          // gfx1250 global_prefetch_b8
        __builtin_prefetch(aptr + k0 + 64, 0, 1);
        __builtin_prefetch(bptr + k0 + 64, 0, 1);
      }
    }

    v16h a = ld_frag(&As[buf][wave * 16][0], 40);
#pragma unroll
    for (int j = 0; j < 8; ++j) {
      v16h b = ld_frag(&Bs[buf][j * 16][0], 40);
      acc[j] = WMMA_F32_F16(a, b, acc[j]);
    }

    if (has_next) wait_async0();       // copy done before buffer swap
    __syncthreads();
  }

  // Epilogue: bias + store. C layout: lane (hi,r) elem i -> (M=hi*8+i, N=r).
#pragma unroll
  for (int j = 0; j < 8; ++j) {
    const int n = n0 + j * 16 + r;
    const float bv = bias[n];
#pragma unroll
    for (int i = 0; i < 8; ++i) {
      const int m = m0 + wave * 16 + hi * 8 + i;
      const float val = acc[j][i] + bv;
      if (OUT_MODE == 0) {
        const int b  = m >> 11;          // / SLEN
        const int s  = m & (SLEN - 1);
        const int h  = n >> 6;           // / HD
        const int dh = n & (HD - 1);
        ((_Float16*)outp)[(((size_t)b * NHEAD + h) * SLEN + s) * HD + dh] =
            (_Float16)val;
      } else {
        ((float*)outp)[(size_t)m * DMODEL + n] = val;
      }
    }
  }
}

// ---------------------------------------------------------------------------
// Flash-style causal attention over f16 head-major Q/K/V.
// Grid: (S/64, B*H). Block: 128 threads (4 waves); wave w owns 16 query rows.
// K-tile staged via async global->LDS; V staged transposed (manual).
// Softmax on all 32 lanes; cross-half combine via __shfl_xor(.,16) (wave32).
// ---------------------------------------------------------------------------
__global__ __launch_bounds__(128)
void attn_wmma(const _Float16* __restrict__ Qp, const _Float16* __restrict__ Kp,
               const _Float16* __restrict__ Vp, _Float16* __restrict__ Ao) {
  __shared__ alignas(16) _Float16 Ks[64][72];      // [key][d]  == B^T for QK^T
  __shared__ alignas(16) _Float16 Vt[64][72];      // [d][key]  == B^T for P*V
  __shared__ alignas(16) _Float16 Ps[4][16][72];   // per-wave probabilities
  __shared__ float Ss[4][16][64];                  // per-wave raw scores
  __shared__ float rowf[4][16];                    // per-row broadcast factor

  const int t    = threadIdx.x;
  const int wave = t >> 5;
  const int lane = t & 31;
  const int r    = lane & 15, hi = lane >> 4;

  const int ib = blockIdx.x;                 // query block (64 rows)
  const int bh = blockIdx.y;                 // b*NHEAD + h
  const int b  = bh >> 4, h = bh & 15;

  const _Float16* Q = Qp + (size_t)bh * SLEN * HD;
  const _Float16* K = Kp + (size_t)bh * SLEN * HD;
  const _Float16* V = Vp + (size_t)bh * SLEN * HD;

  // Q fragments for this wave's 16 rows, K-dim = 64 -> 2 fragments.
  const _Float16* qbase = Q + (size_t)(ib * 64 + wave * 16) * HD;
  const v16h qf0 = ld_frag(qbase, HD);
  const v16h qf1 = ld_frag(qbase + 32, HD);

  v8f oacc[4];
#pragma unroll
  for (int j = 0; j < 4; ++j) oacc[j] = vzero8();

  float m_run = -INFINITY;   // replicated in both half-lanes of each row pair
  float l_run = 0.0f;
  const float scale = 0.022097086912079608f;   // 1/sqrt(MAX_LEN=2048)

  const int ky = t >> 1, c0 = (t & 1) * 32;    // staging assignment

  for (int jb = 0; jb <= ib; ++jb) {
    const int j0 = jb * 64;
    // ---- stage K (async b128 x4, one base pair) and V transposed -------
    {
      const _Float16* kg = K + (size_t)(j0 + ky) * HD + c0;
      _Float16* kl = &Ks[ky][c0];
      async_b128<0>(kg, kl);
      async_b128<16>(kg, kl);
      async_b128<32>(kg, kl);
      async_b128<48>(kg, kl);
      const _Float16* vs = V + (size_t)(j0 + ky) * HD + c0;
#pragma unroll
      for (int d = 0; d < 32; ++d) Vt[c0 + d][ky] = vs[d];
      if (jb < ib) {                         // prefetch next K/V tiles
        __builtin_prefetch(K + (size_t)(j0 + 64 + ky) * HD + c0, 0, 1);
        __builtin_prefetch(V + (size_t)(j0 + 64 + ky) * HD + c0, 0, 1);
      }
      wait_async0();
    }
    __syncthreads();

    // ---- scores: S(16x64) = Q(16x64) . K^T -----------------------------
    v8f sacc[4];
#pragma unroll
    for (int j = 0; j < 4; ++j) sacc[j] = vzero8();
#pragma unroll
    for (int kk = 0; kk < 2; ++kk) {
      const v16h a = kk ? qf1 : qf0;
#pragma unroll
      for (int j4 = 0; j4 < 4; ++j4) {
        v16h bfr = ld_frag(&Ks[j4 * 16][kk * 32], 72);
        sacc[j4] = WMMA_F32_F16(a, bfr, sacc[j4]);
      }
    }
    // scale + causal mask -> LDS
#pragma unroll
    for (int j4 = 0; j4 < 4; ++j4) {
#pragma unroll
      for (int i = 0; i < 8; ++i) {
        const int mrow = hi * 8 + i;
        const int qg = ib * 64 + wave * 16 + mrow;
        const int kg = j0 + j4 * 16 + r;
        float sv = sacc[j4][i] * scale;
        if (kg > qg) sv = -INFINITY;
        Ss[wave][mrow][j4 * 16 + r] = sv;
      }
    }
    __syncthreads();

    // ---- online softmax: lane (hi,r) -> row r, cols hi*32..hi*32+31 ----
    {
      const int cb = hi * 32;
      float mx = -INFINITY;
#pragma unroll 8
      for (int c = 0; c < 32; ++c) mx = fmaxf(mx, Ss[wave][r][cb + c]);
      mx = fmaxf(mx, __shfl_xor(mx, 16, 32));
      const float m_new = fmaxf(m_run, mx);
      const float corr  = __expf(m_run - m_new);   // exp(-inf)=0 first pass
      float psum = 0.0f;
#pragma unroll 8
      for (int c = 0; c < 32; ++c) {
        const float p = __expf(Ss[wave][r][cb + c] - m_new);
        Ps[wave][r][cb + c] = (_Float16)p;
        psum += p;
      }
      psum += __shfl_xor(psum, 16, 32);
      l_run = l_run * corr + psum;
      m_run = m_new;
      if (hi == 0) rowf[wave][r] = corr;
    }
    __syncthreads();

    // ---- rescale accumulator, then O += P . V --------------------------
#pragma unroll
    for (int i = 0; i < 8; ++i) {
      const float c = rowf[wave][hi * 8 + i];
#pragma unroll
      for (int j4 = 0; j4 < 4; ++j4) oacc[j4][i] *= c;
    }
#pragma unroll
    for (int kk = 0; kk < 2; ++kk) {
      v16h a = ld_frag(&Ps[wave][0][kk * 32], 72);
#pragma unroll
      for (int j4 = 0; j4 < 4; ++j4) {
        v16h bfr = ld_frag(&Vt[j4 * 16][kk * 32], 72);
        oacc[j4] = WMMA_F32_F16(a, bfr, oacc[j4]);
      }
    }
    __syncthreads();
  }

  // ---- finalize: O /= l, store merged-head f16 [B][S][H*64] ------------
  if (hi == 0) rowf[wave][r] = 1.0f / l_run;
  __syncthreads();
#pragma unroll
  for (int i = 0; i < 8; ++i) {
    const float inv = rowf[wave][hi * 8 + i];
    const int s = ib * 64 + wave * 16 + hi * 8 + i;
#pragma unroll
    for (int j4 = 0; j4 < 4; ++j4) {
      const int dh = j4 * 16 + r;
      Ao[((size_t)(b * SLEN + s)) * DMODEL + h * HD + dh] =
          (_Float16)(oacc[j4][i] * inv);
    }
  }
}

// ---------------------------------------------------------------------------
extern "C" void kernel_launch(void* const* d_in, const int* in_sizes, int n_in,
                              void* d_out, int out_size, void* d_ws, size_t ws_size,
                              hipStream_t stream) {
  const float* q  = (const float*)d_in[0];
  const float* k  = (const float*)d_in[1];
  const float* v  = (const float*)d_in[2];
  const float* Wq = (const float*)d_in[3];
  const float* bq = (const float*)d_in[4];
  const float* Wk = (const float*)d_in[5];
  const float* bk = (const float*)d_in[6];
  const float* Wv = (const float*)d_in[7];
  const float* bv = (const float*)d_in[8];
  const float* Wf = (const float*)d_in[9];
  const float* bf = (const float*)d_in[10];

  const size_t plane = (size_t)MROWS * DMODEL;    // 8M f16 elements
  const size_t wsz   = (size_t)DMODEL * DMODEL;   // 1M f16 elements
  _Float16* Qp  = (_Float16*)d_ws;
  _Float16* Kp  = Qp + plane;
  _Float16* Vp  = Kp + plane;
  _Float16* Ao  = Vp + plane;
  _Float16* Wqt = Ao + plane;
  _Float16* Wkt = Wqt + wsz;
  _Float16* Wvt = Wkt + wsz;
  _Float16* Wft = Wvt + wsz;
  _Float16* qh  = Wft + wsz;
  _Float16* kh  = qh + plane;
  _Float16* vh  = kh + plane;

  // Activation prep: f16 copies of q/k/v (read 8x each in the GEMMs).
  const int n8 = (int)(plane / 8);
  dim3 cgrid(n8 / 256), cblk(256);
  cvt_f32_f16<<<cgrid, cblk, 0, stream>>>(q, qh, n8);
  cvt_f32_f16<<<cgrid, cblk, 0, stream>>>(k, kh, n8);
  cvt_f32_f16<<<cgrid, cblk, 0, stream>>>(v, vh, n8);

  // Weight prep: f16 transposed copies (reused 64x each).
  dim3 tgrid(DMODEL / 32, DMODEL / 32), tblk(256);
  transpose_f32_f16<<<tgrid, tblk, 0, stream>>>(Wq, Wqt);
  transpose_f32_f16<<<tgrid, tblk, 0, stream>>>(Wk, Wkt);
  transpose_f32_f16<<<tgrid, tblk, 0, stream>>>(Wv, Wvt);
  transpose_f32_f16<<<tgrid, tblk, 0, stream>>>(Wf, Wft);

  dim3 gblk(256), ggrid(DMODEL / 128, MROWS / 128);
  gemm1024_wmma<0><<<ggrid, gblk, 0, stream>>>(qh, Wqt, bq, (void*)Qp);
  gemm1024_wmma<0><<<ggrid, gblk, 0, stream>>>(kh, Wkt, bk, (void*)Kp);
  gemm1024_wmma<0><<<ggrid, gblk, 0, stream>>>(vh, Wvt, bv, (void*)Vp);

  dim3 agrid(SLEN / 64, BATCH * NHEAD), ablk(128);
  attn_wmma<<<agrid, ablk, 0, stream>>>(Qp, Kp, Vp, Ao);

  gemm1024_wmma<1><<<ggrid, gblk, 0, stream>>>(Ao, Wft, bf, d_out);
}
